// DGCNN_16578573762789
// MI455X (gfx1250) — compile-verified, matrix operand order checked by reading, not run on hardware
//
#include <hip/hip_runtime.h>
#include <math.h>

#define KNN   20
#define NPTS  2048
#define BATCH 8
#define EPSV  1e-5f
#define SLOPEV 0.2f

typedef __attribute__((ext_vector_type(16))) _Float16 v16h;
typedef __attribute__((ext_vector_type(8)))  _Float16 v8h;
typedef __attribute__((ext_vector_type(8)))  float    v8f;

__device__ __forceinline__ float lrelu(float z) { return z > 0.f ? z : SLOPEV * z; }

__device__ __forceinline__ v16h cat8(v8h lo, v8h hi) {
    return __builtin_shufflevector(lo, hi, 0, 1, 2, 3, 4, 5, 6, 7,
                                           8, 9, 10, 11, 12, 13, 14, 15);
}

// ---------------------------------------------------------------------------
// WMMA register-tiled GEMM on pre-packed f16 operands:
//   Out[m, o] = sum_k A[m*Kp + k] * W[o*Kp + k]      (f32 accumulate/output)
// One wave32 computes a 32x32 output tile: 2 A-tiles x 2 W-tiles ->
// 4 accumulators, 4 v_wmma_f32_16x16x32_f16 per K-step; each fetched operand
// tile is used twice. Per-lane fragment loads are aligned b128.
// CDNA5 16-bit A/B fragment layout:
//   lanes 0-15 : elems 0-7 -> K 0..7,  elems 8-15 -> K 16..23   (row = lane)
//   lanes16-31 : elems 0-7 -> K 8..15, elems 8-15 -> K 24..31   (row = lane-16)
// C/D f32 layout: lanes 0-15 vgpr j -> (M=j, N=lane); lanes 16-31 -> (M=8+j).
// ---------------------------------------------------------------------------
__global__ __launch_bounds__(32)
void gemm_wmma_kernel(const _Float16* __restrict__ A, const _Float16* __restrict__ W,
                      float* __restrict__ Out, int Kp, int ldo)
{
    const int lane = threadIdx.x;
    const int r    = lane & 15;
    const int hi   = lane >> 4;
    const int o0   = blockIdx.x << 5;
    const int m0   = blockIdx.y << 5;

    const _Float16* ar0 = A + (size_t)(m0 + r)      * Kp;
    const _Float16* ar1 = A + (size_t)(m0 + 16 + r) * Kp;
    const _Float16* wr0 = W + (size_t)(o0 + r)      * Kp;
    const _Float16* wr1 = W + (size_t)(o0 + 16 + r) * Kp;

    v8f acc00 = {}, acc01 = {}, acc10 = {}, acc11 = {};

    for (int kb = 0; kb < Kp; kb += 32) {
        __builtin_prefetch(ar0 + kb + 32, 0, 1);    // global_prefetch_b8
        __builtin_prefetch(wr0 + kb + 32, 0, 1);
        const int klo = kb + hi * 8;
        const int khi = kb + 16 + hi * 8;
        v16h a0 = cat8(*(const v8h*)(ar0 + klo), *(const v8h*)(ar0 + khi));
        v16h a1 = cat8(*(const v8h*)(ar1 + klo), *(const v8h*)(ar1 + khi));
        v16h w0 = cat8(*(const v8h*)(wr0 + klo), *(const v8h*)(wr0 + khi));
        v16h w1 = cat8(*(const v8h*)(wr1 + klo), *(const v8h*)(wr1 + khi));
        // (neg_a, A, neg_b, B, c_mod, C, reuse_a, reuse_b)
        acc00 = __builtin_amdgcn_wmma_f32_16x16x32_f16(false, a0, false, w0, (short)0, acc00, false, false);
        acc01 = __builtin_amdgcn_wmma_f32_16x16x32_f16(false, a0, false, w1, (short)0, acc01, false, false);
        acc10 = __builtin_amdgcn_wmma_f32_16x16x32_f16(false, a1, false, w0, (short)0, acc10, false, false);
        acc11 = __builtin_amdgcn_wmma_f32_16x16x32_f16(false, a1, false, w1, (short)0, acc11, false, false);
    }

    const int mrow = m0 + hi * 8;
    float* out00 = Out + (size_t)mrow * ldo + (o0 + r);
#pragma unroll
    for (int j = 0; j < 8; ++j) {
        out00[(size_t)j * ldo]                       = acc00[j];   // (m0..  , o0..  )
        out00[(size_t)j * ldo + 16]                  = acc01[j];   // (m0..  , o0+16)
        out00[(size_t)(j + 16) * ldo]                = acc10[j];   // (m0+16., o0..  )
        out00[(size_t)(j + 16) * ldo + 16]           = acc11[j];   // (m0+16., o0+16)
    }
}

// ---------------------------------------------------------------------------
// Pack x (B, C, N) channel-major into row-major f16 A [B*N, Kp] (zero-padded),
// optionally producing per-point squared norms xx[b*N+n] (f32).
// ---------------------------------------------------------------------------
__global__ void pack_kernel(const float* __restrict__ x, _Float16* __restrict__ Apk,
                            float* __restrict__ xx, int C, int N, int Kp,
                            int colOff, int zeroPad)
{
    int t = blockIdx.x * blockDim.x + threadIdx.x;      // t = b*N + n
    if (t >= BATCH * N) return;
    int n = t % N;
    int b = t / N;
    _Float16* row = Apk + (size_t)t * Kp;
    if (zeroPad) {
        for (int c = colOff + C; c < Kp; ++c) row[c] = (_Float16)0.f;
    }
    float ss = 0.f;
    for (int c = 0; c < C; ++c) {
        float v = x[((size_t)b * C + c) * N + n];
        row[colOff + c] = (_Float16)v;
        ss += v * v;
    }
    if (xx) xx[t] = ss;
}

// ---------------------------------------------------------------------------
// Per-row top-K over d[n][m] = 2*G[n][m] - xx[n] - xx[m]  (one batch).
// ---------------------------------------------------------------------------
__global__ void topk_kernel(const float* __restrict__ G, const float* __restrict__ xx,
                            int* __restrict__ idx, int N)
{
    int n = blockIdx.x * blockDim.x + threadIdx.x;
    if (n >= N) return;
    float best[KNN];
    int   bi[KNN];
#pragma unroll
    for (int k = 0; k < KNN; ++k) { best[k] = -3.4e38f; bi[k] = 0; }
    const float* grow = G + (size_t)n * N;
    float xn = xx[n];
    for (int m = 0; m < N; ++m) {
        float d = 2.f * grow[m] - xn - xx[m];
        if (d > best[KNN - 1]) {
            int p = KNN - 1;
            while (p > 0 && best[p - 1] < d) {
                best[p] = best[p - 1]; bi[p] = bi[p - 1]; --p;
            }
            best[p] = d; bi[p] = m;
        }
    }
    int* orow = idx + (size_t)n * KNN;
#pragma unroll
    for (int k = 0; k < KNN; ++k) orow[k] = bi[k];
}

// ---------------------------------------------------------------------------
// Split W (O, 2C) into f16 Wa = W[:, :C] and Wd = W[:, C:] - W[:, :C],
// zero-padded to Kp, row-major [O, Kp].
// ---------------------------------------------------------------------------
__global__ void wprep_kernel(const float* __restrict__ W, _Float16* __restrict__ wa,
                             _Float16* __restrict__ wd, int C, int Kp, int total)
{
    int t = blockIdx.x * blockDim.x + threadIdx.x;
    if (t >= total) return;
    int c = t % Kp;
    int o = t / Kp;
    float a = 0.f, d = 0.f;
    if (c < C) {
        a = W[(size_t)o * 2 * C + c];
        d = W[(size_t)o * 2 * C + C + c] - a;
    }
    wa[t] = (_Float16)a;
    wd[t] = (_Float16)d;
}

// f32 -> f16 elementwise (for W5)
__global__ void cvt16_kernel(const float* __restrict__ in, _Float16* __restrict__ out, int total)
{
    int t = blockIdx.x * blockDim.x + threadIdx.x;
    if (t < total) out[t] = (_Float16)in[t];
}

// ---------------------------------------------------------------------------
// Edge-conv epilogue: out[b,o,n] = max_k lrelu(bn(P[b, idx[b,n,k], o] + Q[b,n,o]))
// ---------------------------------------------------------------------------
__global__ void edge_max_kernel(const float* __restrict__ P, const float* __restrict__ Q,
                                const int* __restrict__ idx,
                                const float* __restrict__ g, const float* __restrict__ bb,
                                const float* __restrict__ mm, const float* __restrict__ vv,
                                float* __restrict__ xout, int N, int O)
{
    long t = (long)blockIdx.x * blockDim.x + threadIdx.x;
    long total = (long)BATCH * N * O;
    if (t >= total) return;
    int o = (int)(t % O);
    int n = (int)((t / O) % N);
    int b = (int)(t / ((long)O * N));

    float s  = g[o] * rsqrtf(vv[o] + EPSV);
    float tt = bb[o] - mm[o] * s;
    float q  = Q[((size_t)b * N + n) * O + o];
    const int* ir = idx + ((size_t)b * N + n) * KNN;

    float mx = -3.4e38f;
#pragma unroll
    for (int k = 0; k < KNN; ++k) {
        int m = ir[k];
        float z = P[((size_t)b * N + m) * O + o] + q;
        z = lrelu(z * s + tt);
        mx = fmaxf(mx, z);
    }
    xout[((size_t)b * O + o) * N + n] = mx;
}

// ---------------------------------------------------------------------------
// Global pooling of Y [N, O] (one batch): bn5 + lrelu, then max & mean over N.
// h[o] = gmax, h[O + o] = gavg.  One block per output channel, LDS reduce.
// ---------------------------------------------------------------------------
__global__ void reduce5_kernel(const float* __restrict__ Y,
                               const float* __restrict__ g, const float* __restrict__ bb,
                               const float* __restrict__ mm, const float* __restrict__ vv,
                               float* __restrict__ h, int N, int O)
{
    int o   = blockIdx.x;
    int tid = threadIdx.x;
    float s  = g[o] * rsqrtf(vv[o] + EPSV);
    float tt = bb[o] - mm[o] * s;

    float mx = -3.4e38f, sum = 0.f;
    for (int n = tid; n < N; n += blockDim.x) {
        float z = lrelu(Y[(size_t)n * O + o] * s + tt);
        mx = fmaxf(mx, z);
        sum += z;
    }
    __shared__ float smx[256];
    __shared__ float ssm[256];
    smx[tid] = mx; ssm[tid] = sum;
    __syncthreads();
    for (int st = 128; st > 0; st >>= 1) {
        if (tid < st) {
            smx[tid] = fmaxf(smx[tid], smx[tid + st]);
            ssm[tid] += ssm[tid + st];
        }
        __syncthreads();
    }
    if (tid == 0) {
        h[o]     = smx[0];
        h[O + o] = ssm[0] / (float)N;
    }
}

// ---------------------------------------------------------------------------
// Small FC: out[r,o] = act(bn( in[r,:]·W[o,:] + bias[o] ))
// ---------------------------------------------------------------------------
__global__ void fc_kernel(const float* __restrict__ in, const float* __restrict__ W,
                          const float* __restrict__ bias,
                          const float* __restrict__ g, const float* __restrict__ bb,
                          const float* __restrict__ mm, const float* __restrict__ vv,
                          float* __restrict__ out, int rows, int cin, int cout, int act)
{
    int t = blockIdx.x * blockDim.x + threadIdx.x;
    if (t >= rows * cout) return;
    int o = t % cout;
    int r = t / cout;
    float acc = bias ? bias[o] : 0.f;
    const float* ir = in + (size_t)r * cin;
    const float* wr = W  + (size_t)o * cin;
    for (int c = 0; c < cin; ++c) acc += ir[c] * wr[c];
    if (g) {
        float s = g[o] * rsqrtf(vv[o] + EPSV);
        acc = acc * s + (bb[o] - mm[o] * s);
    }
    if (act) acc = lrelu(acc);
    out[(size_t)r * cout + o] = acc;
}

// ---------------------------------------------------------------------------
extern "C" void kernel_launch(void* const* d_in, const int* in_sizes, int n_in,
                              void* d_out, int out_size, void* d_ws, size_t ws_size,
                              hipStream_t stream)
{
    (void)in_sizes; (void)n_in; (void)out_size; (void)ws_size;

    // --- inputs, setup_inputs() insertion order ---
    const float* x    = (const float*)d_in[0];
    const float* W1   = (const float*)d_in[1];
    const float* bn1g = (const float*)d_in[2],  *bn1b = (const float*)d_in[3],
               * bn1m = (const float*)d_in[4],  *bn1v = (const float*)d_in[5];
    const float* W2   = (const float*)d_in[6];
    const float* bn2g = (const float*)d_in[7],  *bn2b = (const float*)d_in[8],
               * bn2m = (const float*)d_in[9],  *bn2v = (const float*)d_in[10];
    const float* W3   = (const float*)d_in[11];
    const float* bn3g = (const float*)d_in[12], *bn3b = (const float*)d_in[13],
               * bn3m = (const float*)d_in[14], *bn3v = (const float*)d_in[15];
    const float* W4   = (const float*)d_in[16];
    const float* bn4g = (const float*)d_in[17], *bn4b = (const float*)d_in[18],
               * bn4m = (const float*)d_in[19], *bn4v = (const float*)d_in[20];
    const float* W5   = (const float*)d_in[21];
    const float* bn5g = (const float*)d_in[22], *bn5b = (const float*)d_in[23],
               * bn5m = (const float*)d_in[24], *bn5v = (const float*)d_in[25];
    const float* Wl1  = (const float*)d_in[26];
    const float* bn6g = (const float*)d_in[27], *bn6b = (const float*)d_in[28],
               * bn6m = (const float*)d_in[29], *bn6v = (const float*)d_in[30];
    const float* Wl2  = (const float*)d_in[31];
    const float* bl2  = (const float*)d_in[32];
    const float* bn7g = (const float*)d_in[33], *bn7b = (const float*)d_in[34],
               * bn7m = (const float*)d_in[35], *bn7v = (const float*)d_in[36];
    const float* Wl3  = (const float*)d_in[37];
    const float* bl3  = (const float*)d_in[38];

    // --- workspace layout (~110 MB, L2-resident working set per stage) ---
    char* ws = (char*)d_ws;
    const size_t MB = 1ull << 20;
    _Float16* APK = (_Float16*)(ws + 0);          // [16384 x 512] f16 activations (16 MB)
    float*    GRAM= (float*)(ws + 16 * MB);       // [2048 x 2048] per-batch Gram (16 MB)
    float*    Pm  = (float*)(ws + 32 * MB);       // [16384 x O] neighbor term (16 MB)
    float*    Qm  = (float*)(ws + 48 * MB);       // [16384 x O] center term  (16 MB)
    float*    X1  = (float*)(ws + 64 * MB);       // (8, 64,2048)
    float*    X2  = (float*)(ws + 68 * MB);       // (8, 64,2048)
    float*    X3  = (float*)(ws + 72 * MB);       // (8,128,2048)
    float*    X4  = (float*)(ws + 80 * MB);       // (8,256,2048)
    float*    Y5  = (float*)(ws + 96 * MB);       // [2048 x 1024] per-batch (8 MB)
    float*    XX  = (float*)(ws + 104 * MB);      // [16384] sq norms
    int*      IDX = (int*)  (ws + 105 * MB);      // [16384 x 20]
    _Float16* WAh = (_Float16*)(ws + 107 * MB);             // [O x Kp] f16
    _Float16* WDh = (_Float16*)(ws + 107 * MB + 128 * 1024);
    _Float16* W5h = (_Float16*)(ws + 108 * MB);   // [1024 x 512] f16 (1 MB)
    float*    H   = (float*)(ws + 109 * MB);      // (8, 2048)
    float*    H1  = (float*)(ws + 109 * MB + 128 * 1024); // (8, 512)
    float*    H2  = (float*)(ws + 109 * MB + 256 * 1024); // (8, 256)

    const int BN = BATCH * NPTS;                  // 16384 rows

    // --- 4 edge-conv layers ---
    const float* srcs[4] = { x, X1, X2, X3 };
    float*       outs[4] = { X1, X2, X3, X4 };
    const int    Cs[4]  = { 5, 64, 64, 128 };
    const int    Kps[4] = { 32, 64, 64, 128 };    // K padded to multiple of 32
    const int    Os[4]  = { 64, 64, 128, 256 };
    const float* Wm[4]  = { W1, W2, W3, W4 };
    const float* Gg[4]  = { bn1g, bn2g, bn3g, bn4g };
    const float* Gb[4]  = { bn1b, bn2b, bn3b, bn4b };
    const float* Gm[4]  = { bn1m, bn2m, bn3m, bn4m };
    const float* Gv[4]  = { bn1v, bn2v, bn3v, bn4v };

    for (int l = 0; l < 4; ++l) {
        const int C = Cs[l], Kp = Kps[l], O = Os[l];

        // pack X^T [B*N, Kp] f16 + squared norms
        pack_kernel<<<BN / 256, 256, 0, stream>>>(srcs[l], APK, XX, C, NPTS, Kp, 0, 1);

        // kNN: per-batch WMMA Gram + register top-20
        for (int b = 0; b < BATCH; ++b) {
            const _Float16* Ab = APK + (size_t)b * NPTS * Kp;
            gemm_wmma_kernel<<<dim3(NPTS / 32, NPTS / 32), 32, 0, stream>>>(
                Ab, Ab, GRAM, Kp, NPTS);
            topk_kernel<<<NPTS / 256, 256, 0, stream>>>(
                GRAM, XX + b * NPTS, IDX + (size_t)b * NPTS * KNN, NPTS);
        }

        // split weights: Wa, Wd = Wb - Wa (f16, zero-padded to Kp)
        wprep_kernel<<<(O * Kp + 255) / 256, 256, 0, stream>>>(Wm[l], WAh, WDh, C, Kp, O * Kp);

        // P = X·Wa^T, Q = X·(Wb-Wa)^T  (dense WMMA GEMMs over all 16384 points)
        gemm_wmma_kernel<<<dim3(O / 32, BN / 32), 32, 0, stream>>>(APK, WAh, Pm, Kp, O);
        gemm_wmma_kernel<<<dim3(O / 32, BN / 32), 32, 0, stream>>>(APK, WDh, Qm, Kp, O);

        // gather + bn + lrelu + max over 20 neighbors
        long tot = (long)BN * O;
        edge_max_kernel<<<(int)((tot + 255) / 256), 256, 0, stream>>>(
            Pm, Qm, IDX, Gg[l], Gb[l], Gm[l], Gv[l], outs[l], NPTS, O);
    }

    // --- concat x1..x4 -> [B*N, 512] f16, W5 GEMM, bn5+lrelu, global max/mean ---
    pack_kernel<<<BN / 256, 256, 0, stream>>>(X1, APK, nullptr,  64, NPTS, 512,   0, 0);
    pack_kernel<<<BN / 256, 256, 0, stream>>>(X2, APK, nullptr,  64, NPTS, 512,  64, 0);
    pack_kernel<<<BN / 256, 256, 0, stream>>>(X3, APK, nullptr, 128, NPTS, 512, 128, 0);
    pack_kernel<<<BN / 256, 256, 0, stream>>>(X4, APK, nullptr, 256, NPTS, 512, 256, 0);
    cvt16_kernel<<<(1024 * 512 + 255) / 256, 256, 0, stream>>>(W5, W5h, 1024 * 512);

    for (int b = 0; b < BATCH; ++b) {
        gemm_wmma_kernel<<<dim3(1024 / 32, NPTS / 32), 32, 0, stream>>>(
            APK + (size_t)b * NPTS * 512, W5h, Y5, 512, 1024);
        reduce5_kernel<<<1024, 256, 0, stream>>>(
            Y5, bn5g, bn5b, bn5m, bn5v, H + (size_t)b * 2048, NPTS, 1024);
    }

    // --- FC head ---
    fc_kernel<<<(8 * 512 + 255) / 256, 256, 0, stream>>>(
        H, Wl1, nullptr, bn6g, bn6b, bn6m, bn6v, H1, 8, 2048, 512, 1);
    fc_kernel<<<(8 * 256 + 255) / 256, 256, 0, stream>>>(
        H1, Wl2, bl2, bn7g, bn7b, bn7m, bn7v, H2, 8, 512, 256, 1);
    fc_kernel<<<1, 256, 0, stream>>>(
        H2, Wl3, bl3, nullptr, nullptr, nullptr, nullptr,
        (float*)d_out, 8, 256, 1, 0);
}
